// DiT_47442208752042
// MI455X (gfx1250) — matrix operand order repaired and checked by reading, not compiled, not run on hardware
//
#include <hip/hip_runtime.h>
#include <hip/hip_bf16.h>
#include <stdint.h>

// ---------------------------------------------------------------------------
// Types for CDNA5 WMMA
// ---------------------------------------------------------------------------
typedef __attribute__((ext_vector_type(16))) __bf16 v16bf;
typedef __attribute__((ext_vector_type(8)))  float  v8f;

__device__ inline unsigned short f32_to_bf16(float f) {
    union { float f; unsigned int u; } c; c.f = f;
    unsigned int u = c.u;
    u += 0x7FFFu + ((u >> 16) & 1u);   // round-to-nearest-even
    return (unsigned short)(u >> 16);
}

__device__ inline v8f wmma_bf16(v16bf a, v16bf b, v8f c) {
    return __builtin_amdgcn_wmma_f32_16x16x32_bf16(
        /*neg_a=*/false, a, /*neg_b=*/false, b,
        /*c_mod=*/(short)0, c, /*reuse_a=*/false, /*reuse_b=*/false);
}

union Frag16 { unsigned int u[8]; v16bf v; };

// ---------------------------------------------------------------------------
// CDNA5 async global->LDS copy (16B per lane) + asynccnt wait
// Builtin signature (from clang diagnostic): (int4 AS1*, int4 AS3*, imm, imm)
// ---------------------------------------------------------------------------
typedef int v4i __attribute__((vector_size(16)));
typedef __attribute__((address_space(1))) v4i* as1_v4i_ptr;
typedef __attribute__((address_space(3))) v4i* as3_v4i_ptr;

__device__ inline void async_b128(void* lds, const void* g) {
#if __has_builtin(__builtin_amdgcn_global_load_async_to_lds_b128)
    void* gq = (void*)g;   // drop const, then addrspacecast generic->AS1/AS3
    __builtin_amdgcn_global_load_async_to_lds_b128(
        (as1_v4i_ptr)gq, (as3_v4i_ptr)lds, 0, 0);
#else
    unsigned loff = (unsigned)(size_t)(__attribute__((address_space(3))) void*)lds;
    asm volatile("global_load_async_to_lds_b128 %0, %1, off"
                 :: "v"(loff), "v"(g) : "memory");
#endif
}

__device__ inline void async_wait0() {
#if __has_builtin(__builtin_amdgcn_s_wait_asynccnt)
    __builtin_amdgcn_s_wait_asynccnt(0);
#else
    asm volatile("s_wait_asynccnt 0x0" ::: "memory");
#endif
}

// ---------------------------------------------------------------------------
// f32 -> bf16 conversion with row padding (pads zero-filled)
// dst[r][kp] = (k < K) ? bf16(src[r][K]) : 0
// ---------------------------------------------------------------------------
__global__ __launch_bounds__(256)
void cvt_bf16_kernel(const float* __restrict__ src, unsigned short* __restrict__ dst,
                     int K, int Kp, long total /* rows*Kp */)
{
    const long idx = (long)blockIdx.x * 256 + threadIdx.x;
    if (idx >= total) return;
    const int k = (int)(idx % Kp);
    const long r = idx / Kp;
    dst[idx] = (k < K) ? f32_to_bf16(src[r * K + k]) : (unsigned short)0;
}

// ---------------------------------------------------------------------------
// GEMM:  Y[M,N] = A[M,K] @ W[N,K]^T (+bias[n]);  A,W bf16 (padded rows), Y f32
// 128 threads (4 waves), 64x64 tile, K-step 32. Tiles staged with async
// load-to-LDS (full tiles) or guarded scalar copies (N edge).
// Kp must be a multiple of 32 with zeroed pads; lda/ldb row pitches (elems).
// ---------------------------------------------------------------------------
#define LDA_PAD 40   // 80B rows: 16B-aligned, conflict-free frag reads

__global__ __launch_bounds__(128)
void gemm_bf16_kernel(const unsigned short* __restrict__ A,
                      const unsigned short* __restrict__ W,
                      const float* __restrict__ bias, float* __restrict__ Y,
                      int M, int N, int Kp, int lda, int ldb)
{
    __shared__ unsigned short As[64 * LDA_PAD];
    __shared__ unsigned short Bs[64 * LDA_PAD];

    const int tid  = threadIdx.x;
    const int wave = tid >> 5;
    const int lane = tid & 31;
    const int lr   = lane & 15;
    const int lh   = lane >> 4;

    const int m0 = blockIdx.y * 64;
    const int n0 = blockIdx.x * 64;
    const int wm = (wave >> 1) * 32;
    const int wn = (wave & 1) * 32;
    const bool fullN = (n0 + 64 <= N);

    v8f acc[2][2];
    #pragma unroll
    for (int i = 0; i < 2; ++i)
        #pragma unroll
        for (int j = 0; j < 2; ++j)
            acc[i][j] = v8f{};

    const int kTiles = Kp >> 5;
    for (int kt = 0; kt < kTiles; ++kt) {
        const int k0 = kt << 5;
        // ---- stage A (always full: M=mult of 64, Kp padded): async copies
        #pragma unroll
        for (int c = 0; c < 2; ++c) {
            const int ch = tid * 2 + c;          // 0..255
            const int r  = ch >> 2, cb = (ch & 3) * 8;
            async_b128(&As[r * LDA_PAD + cb], A + (size_t)(m0 + r) * lda + k0 + cb);
        }
        // ---- stage B
        if (fullN) {
            #pragma unroll
            for (int c = 0; c < 2; ++c) {
                const int ch = tid * 2 + c;
                const int r  = ch >> 2, cb = (ch & 3) * 8;
                async_b128(&Bs[r * LDA_PAD + cb], W + (size_t)(n0 + r) * ldb + k0 + cb);
            }
        } else {
            const int r = tid >> 1, cb0 = (tid & 1) * 16;
            const int gn = n0 + r;
            #pragma unroll
            for (int c = 0; c < 16; ++c)
                Bs[r * LDA_PAD + cb0 + c] =
                    (gn < N) ? W[(size_t)gn * ldb + k0 + cb0 + c] : (unsigned short)0;
        }
        if (kt + 1 < kTiles)
            __builtin_prefetch(A + (size_t)(m0 + (tid >> 1)) * lda + k0 + 32, 0, 1);
        async_wait0();
        __syncthreads();

        #pragma unroll
        for (int i = 0; i < 2; ++i) {
            Frag16 fa;
            const int am = wm + i * 16 + lr;
            #pragma unroll
            for (int g = 0; g < 8; ++g) {
                // A-matrix 16-bit 16x32 layout: lanes 0-15 k=0..7,16..23; lanes 16-31 +8
                const int k = (g < 4) ? (lh * 8 + 2 * g) : (16 + lh * 8 + 2 * (g - 4));
                fa.u[g] = *(const unsigned int*)&As[am * LDA_PAD + k];
            }
            #pragma unroll
            for (int j = 0; j < 2; ++j) {
                Frag16 fb;
                const int bn = wn + j * 16 + lr;
                #pragma unroll
                for (int g = 0; g < 8; ++g) {
                    // B-matrix 32x16: lanes 0-15 k=0..7,16..23; lanes 16-31 +8
                    const int k = (g < 4) ? (lh * 8 + 2 * g) : (16 + lh * 8 + 2 * (g - 4));
                    fb.u[g] = *(const unsigned int*)&Bs[bn * LDA_PAD + k];
                }
                acc[i][j] = wmma_bf16(fa.v, fb.v, acc[i][j]);
            }
        }
        __syncthreads();
    }

    // C/D layout: VGPR g -> row lh*8+g ; col = lr
    #pragma unroll
    for (int i = 0; i < 2; ++i) {
        #pragma unroll
        for (int j = 0; j < 2; ++j) {
            const int n = n0 + wn + j * 16 + lr;
            if (n >= N) continue;
            const float bv = bias ? bias[n] : 0.0f;
            #pragma unroll
            for (int g = 0; g < 8; ++g) {
                const int m = m0 + wm + i * 16 + lh * 8 + g;
                Y[(size_t)m * N + n] = acc[i][j][g] + bv;
            }
        }
    }
}

// ---------------------------------------------------------------------------
// Fused attention, one workgroup (128 thr = 4 waves) per (b,h).
// q/k/v are bf16 [b,h,256,64]. K staged via async copies, V^T via transpose
// copy; scores in registers; register softmax; probs bf16 in LDS -> P@V WMMA.
// Output written bf16 (feeds proj GEMM).
// ---------------------------------------------------------------------------
#define ATT_LDK 72    // 64 + 8 pad (u16) -> 144B rows (16B aligned)
#define ATT_LDV 264   // 256 + 8 pad (u16)
#define ATT_LDP 264
#define ATT_SMEM ((256 * ATT_LDK + 64 * ATT_LDV + 4 * 16 * ATT_LDP) * 2)

__global__ __launch_bounds__(128)
void attention_kernel(const unsigned short* __restrict__ qb,
                      const unsigned short* __restrict__ kb,
                      const unsigned short* __restrict__ vb,
                      unsigned short* __restrict__ o)
{
    extern __shared__ unsigned short smem[];
    unsigned short* Ks = smem;                       // [256][ATT_LDK]
    unsigned short* Vt = Ks + 256 * ATT_LDK;         // [64][ATT_LDV]  (d-major)
    unsigned short* Pb = Vt + 64 * ATT_LDV;          // [4 waves][16][ATT_LDP]

    const int bh  = blockIdx.x;           // b*16 + h
    const int b   = bh >> 4, h = bh & 15;
    const int tid = threadIdx.x, wave = tid >> 5, lane = tid & 31;
    const int lr  = lane & 15, lh = lane >> 4;

    const unsigned short* kptr = kb + (size_t)bh * 256 * 64;
    const unsigned short* vptr = vb + (size_t)bh * 256 * 64;
    const unsigned short* qptr = qb + (size_t)bh * 256 * 64;

    // K: straight async copy (8 x 16B chunks per row)
    for (int ch = tid; ch < 2048; ch += 128) {
        const int n = ch >> 3, c8 = (ch & 7) * 8;
        async_b128(&Ks[n * ATT_LDK + c8], kptr + (size_t)n * 64 + c8);
    }
    // V^T: vector global read, scattered LDS writes
    for (int ch = tid; ch < 2048; ch += 128) {
        const int n = ch >> 3, d0 = (ch & 7) * 8;
        union { uint4 u4; unsigned short s[8]; } vv;
        vv.u4 = *(const uint4*)(vptr + (size_t)n * 64 + d0);
        #pragma unroll
        for (int i = 0; i < 8; ++i) Vt[(d0 + i) * ATT_LDV + n] = vv.s[i];
    }
    async_wait0();
    __syncthreads();

    unsigned short* P = Pb + wave * 16 * ATT_LDP;

    for (int qt = wave; qt < 16; qt += 4) {
        const int qbase = qt * 16;
        // Q fragments (2 k-halves of HD=64), dword pair loads from global bf16
        Frag16 qf[2];
        #pragma unroll
        for (int f = 0; f < 2; ++f) {
            const int qm = qbase + lr;
            #pragma unroll
            for (int g = 0; g < 8; ++g) {
                const int k = f * 32 + ((g < 4) ? (lh * 8 + 2 * g)
                                                : (16 + lh * 8 + 2 * (g - 4)));
                qf[f].u[g] = *(const unsigned int*)&qptr[(size_t)qm * 64 + k];
            }
        }
        // scores: 16 key tiles of 16
        v8f sc[16];
        #pragma unroll
        for (int t = 0; t < 16; ++t) {
            v8f a = v8f{};
            #pragma unroll
            for (int f = 0; f < 2; ++f) {
                Frag16 kf;
                const int kn = t * 16 + lr;
                #pragma unroll
                for (int g = 0; g < 8; ++g) {
                    const int k = f * 32 + ((g < 4) ? (lh * 8 + 2 * g) : (16 + lh * 8 + 2 * (g - 4)));
                    kf.u[g] = *(const unsigned int*)&Ks[kn * ATT_LDK + k];
                }
                a = wmma_bf16(qf[f].v, kf.v, a);
            }
            sc[t] = a * 0.125f;     // HD^-0.5
        }
        // register softmax: row r = lh*8+g, cols {t*16+lr}
        float inv_sum[8];
        #pragma unroll
        for (int g = 0; g < 8; ++g) {
            float m = -1e30f;
            #pragma unroll
            for (int t = 0; t < 16; ++t) m = fmaxf(m, sc[t][g]);
            #pragma unroll
            for (int s = 1; s < 16; s <<= 1) m = fmaxf(m, __shfl_xor(m, s, 32));
            float ssum = 0.0f;
            #pragma unroll
            for (int t = 0; t < 16; ++t) {
                const float e = __expf(sc[t][g] - m);
                sc[t][g] = e; ssum += e;
            }
            #pragma unroll
            for (int s = 1; s < 16; s <<= 1) ssum += __shfl_xor(ssum, s, 32);
            inv_sum[g] = 1.0f / ssum;
        }
        #pragma unroll
        for (int t = 0; t < 16; ++t)
            #pragma unroll
            for (int g = 0; g < 8; ++g)
                P[(lh * 8 + g) * ATT_LDP + t * 16 + lr] =
                    f32_to_bf16(sc[t][g] * inv_sum[g]);

        // O = P @ V   (A = P[16x256] bf16, B = Vt rows over d)
        v8f oacc[4];
        #pragma unroll
        for (int nd = 0; nd < 4; ++nd) oacc[nd] = v8f{};
        #pragma unroll
        for (int kt = 0; kt < 8; ++kt) {
            Frag16 pf;
            #pragma unroll
            for (int g = 0; g < 8; ++g) {
                const int k = kt * 32 + ((g < 4) ? (lh * 8 + 2 * g)
                                                 : (16 + lh * 8 + 2 * (g - 4)));
                pf.u[g] = *(const unsigned int*)&P[lr * ATT_LDP + k];
            }
            #pragma unroll
            for (int nd = 0; nd < 4; ++nd) {
                Frag16 vf;
                const int vn = nd * 16 + lr;
                #pragma unroll
                for (int g = 0; g < 8; ++g) {
                    const int k = kt * 32 + ((g < 4) ? (lh * 8 + 2 * g) : (16 + lh * 8 + 2 * (g - 4)));
                    vf.u[g] = *(const unsigned int*)&Vt[vn * ATT_LDV + k];
                }
                oacc[nd] = wmma_bf16(pf.v, vf.v, oacc[nd]);
            }
        }
        // store bf16 to [B,N,C] with c = h*64+d
        #pragma unroll
        for (int nd = 0; nd < 4; ++nd) {
            const int d = nd * 16 + lr;
            #pragma unroll
            for (int g = 0; g < 8; ++g) {
                const int q = qbase + lh * 8 + g;
                o[((size_t)(b * 256 + q)) * 1024 + h * 64 + d] = f32_to_bf16(oacc[nd][g]);
            }
        }
    }
}

// ---------------------------------------------------------------------------
// Patch embed + pos embed (K=16, scalar)
// ---------------------------------------------------------------------------
__global__ __launch_bounds__(256)
void patch_embed_kernel(const float* __restrict__ x, const float* __restrict__ pw,
                        const float* __restrict__ pb, const float* __restrict__ pos,
                        float* __restrict__ tok)
{
    const int bn = blockIdx.x;                // b*256 + n
    const int b = bn >> 8, n = bn & 255;
    const int hy = n >> 4, wx = n & 15;
    __shared__ float patch[16];
    const int t = threadIdx.x;
    if (t < 16) {
        const int ci = t >> 2, py = (t >> 1) & 1, px = t & 1;
        patch[t] = x[(((size_t)b * 4 + ci) * 32 + hy * 2 + py) * 32 + wx * 2 + px];
    }
    __syncthreads();
    for (int c = t; c < 1024; c += 256) {
        float acc = pb[c] + pos[(size_t)n * 1024 + c];
        const float* wr = pw + (size_t)c * 16;
        #pragma unroll
        for (int k = 0; k < 16; ++k) acc += patch[k] * wr[k];
        tok[(size_t)bn * 1024 + c] = acc;
    }
}

// ---------------------------------------------------------------------------
// Conditioning: silu_c = silu(class_embed[y] + silu(cfg*w1+b1) @ w2^T + b2)
// ---------------------------------------------------------------------------
__global__ __launch_bounds__(256)
void cond_kernel(const int* __restrict__ y, const float* __restrict__ cfg,
                 const float* __restrict__ ce, const float* __restrict__ w1,
                 const float* __restrict__ b1, const float* __restrict__ w2,
                 const float* __restrict__ b2, float* __restrict__ silu_c)
{
    __shared__ float hc[1024];
    const int b = blockIdx.x, t = threadIdx.x;
    const float cs = cfg[b];
    for (int j = t; j < 1024; j += 256) {
        const float hv = cs * w1[j] + b1[j];
        hc[j] = hv / (1.0f + __expf(-hv));
    }
    __syncthreads();
    const int cls = y[b];
    for (int i = t; i < 1024; i += 256) {
        float acc = ce[(size_t)cls * 1024 + i] + b2[i];
        const float* wr = w2 + (size_t)i * 1024;
        for (int j = 0; j < 1024; ++j) acc += hc[j] * wr[j];
        silu_c[b * 1024 + i] = acc / (1.0f + __expf(-acc));
    }
}

// ---------------------------------------------------------------------------
// vec @ W^T + b for adaLN mods
// ---------------------------------------------------------------------------
__global__ __launch_bounds__(256)
void vecmat_kernel(const float* __restrict__ vin, const float* __restrict__ W,
                   const float* __restrict__ bias, float* __restrict__ out,
                   int Bb, int Nout, int K)
{
    const int r = blockIdx.y / Bb, b = blockIdx.y % Bb;
    const int i = blockIdx.x * 256 + threadIdx.x;
    if (i >= Nout) return;
    const float* v = vin + (size_t)b * K;
    const float* w = W + ((size_t)r * Nout + i) * K;
    float acc = bias ? bias[(size_t)r * Nout + i] : 0.0f;
    for (int c = 0; c < K; ++c) acc += v[c] * w[c];
    out[((size_t)r * Bb + b) * Nout + i] = acc;
}

// ---------------------------------------------------------------------------
// xm(bf16) = rmsnorm(xc, nw) * (1 + mods[b][scale_off+c]) + mods[b][shift_off+c]
// ---------------------------------------------------------------------------
__global__ __launch_bounds__(256)
void modulate_rms_kernel(const float* __restrict__ xc, const float* __restrict__ nw,
                         const float* __restrict__ mods_l, int shift_off, int scale_off,
                         unsigned short* __restrict__ xm)
{
    const int bn = blockIdx.x, b = bn >> 8;
    const int t = threadIdx.x;
    const float* xr = xc + (size_t)bn * 1024;
    float vl[4]; float ss = 0.0f;
    #pragma unroll
    for (int i = 0; i < 4; ++i) { const float v = xr[t * 4 + i]; vl[i] = v; ss += v * v; }
    #pragma unroll
    for (int s = 1; s < 32; s <<= 1) ss += __shfl_xor(ss, s, 32);
    __shared__ float red[8];
    if ((t & 31) == 0) red[t >> 5] = ss;
    __syncthreads();
    float tot = 0.0f;
    #pragma unroll
    for (int w = 0; w < 8; ++w) tot += red[w];
    const float r = rsqrtf(tot / 1024.0f + 1e-6f);
    const float* mrow = mods_l + (size_t)b * 6144;
    unsigned short res[4];
    #pragma unroll
    for (int i = 0; i < 4; ++i) {
        const int c = t * 4 + i;
        res[i] = f32_to_bf16(vl[i] * r * nw[c] * (1.0f + mrow[scale_off + c]) +
                             mrow[shift_off + c]);
    }
    unsigned short* dst = xm + (size_t)bn * 1024 + t * 4;
    *(unsigned int*)&dst[0] = (unsigned int)res[0] | ((unsigned int)res[1] << 16);
    *(unsigned int*)&dst[2] = (unsigned int)res[2] | ((unsigned int)res[3] << 16);
}

// ---------------------------------------------------------------------------
// QKV post: q/k head-RMSNorm + RoPE; emit bf16 q,k,v in [b,h,n,d]
// ---------------------------------------------------------------------------
__global__ __launch_bounds__(256)
void qkv_post_kernel(const float* __restrict__ qkv, const float* __restrict__ qnw,
                     const float* __restrict__ knw, unsigned short* __restrict__ qo,
                     unsigned short* __restrict__ ko, unsigned short* __restrict__ vo)
{
    const int bn = blockIdx.x, b = bn >> 8, n = bn & 255;
    const int t = threadIdx.x;
    const int h = t >> 4, g = t & 15;
    const int d0 = g * 4;
    const float* row = qkv + (size_t)bn * 3072;
    float q[4], k[4], v[4];
    #pragma unroll
    for (int i = 0; i < 4; ++i) {
        q[i] = row[h * 64 + d0 + i];
        k[i] = row[1024 + h * 64 + d0 + i];
        v[i] = row[2048 + h * 64 + d0 + i];
    }
    float sq = 0.0f, sk = 0.0f;
    #pragma unroll
    for (int i = 0; i < 4; ++i) { sq += q[i] * q[i]; sk += k[i] * k[i]; }
    #pragma unroll
    for (int s = 1; s < 16; s <<= 1) { sq += __shfl_xor(sq, s, 32); sk += __shfl_xor(sk, s, 32); }
    const float rq = rsqrtf(sq / 64.0f + 1e-6f);
    const float rk = rsqrtf(sk / 64.0f + 1e-6f);
    #pragma unroll
    for (int i = 0; i < 4; ++i) { q[i] *= rq * qnw[d0 + i]; k[i] *= rk * knw[d0 + i]; }
    float qp[4], kp[4];
    #pragma unroll
    for (int i = 0; i < 4; ++i) { qp[i] = __shfl_xor(q[i], 8, 32); kp[i] = __shfl_xor(k[i], 8, 32); }
    #pragma unroll
    for (int i = 0; i < 4; ++i) {
        const int d = d0 + i, j = d & 31;
        const float ang = (float)n * __expf(-(float)(2 * j) * (9.210340371976184f / 64.0f));
        const float c = __cosf(ang), s = __sinf(ang);
        const float rhq = (d < 32) ? -qp[i] : qp[i];
        const float rhk = (d < 32) ? -kp[i] : kp[i];
        q[i] = q[i] * c + rhq * s;
        k[i] = k[i] * c + rhk * s;
    }
    const size_t base = (((size_t)b * 16 + h) * 256 + n) * 64 + d0;
    *(unsigned int*)&qo[base]     = (unsigned int)f32_to_bf16(q[0]) | ((unsigned int)f32_to_bf16(q[1]) << 16);
    *(unsigned int*)&qo[base + 2] = (unsigned int)f32_to_bf16(q[2]) | ((unsigned int)f32_to_bf16(q[3]) << 16);
    *(unsigned int*)&ko[base]     = (unsigned int)f32_to_bf16(k[0]) | ((unsigned int)f32_to_bf16(k[1]) << 16);
    *(unsigned int*)&ko[base + 2] = (unsigned int)f32_to_bf16(k[2]) | ((unsigned int)f32_to_bf16(k[3]) << 16);
    *(unsigned int*)&vo[base]     = (unsigned int)f32_to_bf16(v[0]) | ((unsigned int)f32_to_bf16(v[1]) << 16);
    *(unsigned int*)&vo[base + 2] = (unsigned int)f32_to_bf16(v[2]) | ((unsigned int)f32_to_bf16(v[3]) << 16);
}

// ---------------------------------------------------------------------------
// xc += gate[b][c] * y
// ---------------------------------------------------------------------------
__global__ __launch_bounds__(256)
void residual_kernel(float* __restrict__ xc, const float* __restrict__ y,
                     const float* __restrict__ mods_l, int gate_off, int total)
{
    const int idx = blockIdx.x * 256 + threadIdx.x;
    if (idx >= total) return;
    const int c = idx & 1023, b = idx >> 18;  // 256*1024 per batch
    xc[idx] += mods_l[(size_t)b * 6144 + gate_off + c] * y[idx];
}

// ---------------------------------------------------------------------------
// swiglu: out(bf16,[M][Swp]) = silu(h1)*h3 ; pads zero-filled
// ---------------------------------------------------------------------------
__global__ __launch_bounds__(256)
void swiglu_bf16_kernel(const float* __restrict__ h1, const float* __restrict__ h3,
                        unsigned short* __restrict__ out, int Sw, int Swp, long total)
{
    const long idx = (long)blockIdx.x * 256 + threadIdx.x;
    if (idx >= total) return;
    const int col = (int)(idx % Swp);
    const long row = idx / Swp;
    if (col < Sw) {
        const float a = h1[row * Sw + col];
        out[idx] = f32_to_bf16((a / (1.0f + __expf(-a))) * h3[row * Sw + col]);
    } else {
        out[idx] = 0;
    }
}

// ---------------------------------------------------------------------------
// Final LayerNorm (no affine) + modulate with finmod = [shift | scale] -> f32
// ---------------------------------------------------------------------------
__global__ __launch_bounds__(256)
void final_mod_kernel(const float* __restrict__ xc, const float* __restrict__ fm,
                      float* __restrict__ xm)
{
    const int bn = blockIdx.x, b = bn >> 8;
    const int t = threadIdx.x;
    const float* xr = xc + (size_t)bn * 1024;
    float vl[4]; float s = 0.0f;
    #pragma unroll
    for (int i = 0; i < 4; ++i) { vl[i] = xr[t * 4 + i]; s += vl[i]; }
    #pragma unroll
    for (int k = 1; k < 32; k <<= 1) s += __shfl_xor(s, k, 32);
    __shared__ float red[8];
    if ((t & 31) == 0) red[t >> 5] = s;
    __syncthreads();
    float tot = 0.0f;
    #pragma unroll
    for (int w = 0; w < 8; ++w) tot += red[w];
    const float mean = tot / 1024.0f;
    __syncthreads();
    float vs = 0.0f;
    #pragma unroll
    for (int i = 0; i < 4; ++i) { const float d = vl[i] - mean; vs += d * d; }
    #pragma unroll
    for (int k = 1; k < 32; k <<= 1) vs += __shfl_xor(vs, k, 32);
    if ((t & 31) == 0) red[t >> 5] = vs;
    __syncthreads();
    float tv = 0.0f;
    #pragma unroll
    for (int w = 0; w < 8; ++w) tv += red[w];
    const float r = rsqrtf(tv / 1024.0f + 1e-5f);
    const float* mrow = fm + (size_t)b * 2048;
    #pragma unroll
    for (int i = 0; i < 4; ++i) {
        const int c = t * 4 + i;
        xm[(size_t)bn * 1024 + c] = (vl[i] - mean) * r * (1.0f + mrow[1024 + c]) + mrow[c];
    }
}

// ---------------------------------------------------------------------------
// Final linear (N=16) fused with unpatchify -> d_out [B,OC,32,32]
// ---------------------------------------------------------------------------
__global__ __launch_bounds__(256)
void final_linear_kernel(const float* __restrict__ xm, const float* __restrict__ lw,
                         const float* __restrict__ lb, float* __restrict__ out)
{
    const int bn = blockIdx.x, b = bn >> 8, n = bn & 255;
    const int t = threadIdx.x;
    const int j = t >> 4, part = t & 15;
    const float* xr = xm + (size_t)bn * 1024;
    const float* wr = lw + (size_t)j * 1024;
    float acc = 0.0f;
    for (int c = part; c < 1024; c += 16) acc += xr[c] * wr[c];
    #pragma unroll
    for (int s = 1; s < 16; s <<= 1) acc += __shfl_xor(acc, s, 32);
    if (part == 0) {
        acc += lb[j];
        const int py = j >> 3, px = (j >> 2) & 1, oc = j & 3;
        const int hy = n >> 4, wx = n & 15;
        const int yy = hy * 2 + py, xx = wx * 2 + px;
        out[(((size_t)b * 4 + oc) * 32 + yy) * 32 + xx] = acc;
    }
}

// ---------------------------------------------------------------------------
// Host orchestration
// ---------------------------------------------------------------------------
static inline int cdiv(long a, long b) { return (int)((a + b - 1) / b); }

extern "C" void kernel_launch(void* const* d_in, const int* in_sizes, int n_in,
                              void* d_out, int out_size, void* d_ws, size_t ws_size,
                              hipStream_t stream)
{
    const float* x          = (const float*)d_in[0];
    const int*   y          = (const int*)  d_in[1];
    const float* cfg_scale  = (const float*)d_in[2];
    const float* patch_w    = (const float*)d_in[3];
    const float* patch_b    = (const float*)d_in[4];
    const float* pos_embed  = (const float*)d_in[5];
    const float* class_emb  = (const float*)d_in[6];
    const float* cfg_w1     = (const float*)d_in[7];
    const float* cfg_b1     = (const float*)d_in[8];
    const float* cfg_w2     = (const float*)d_in[9];
    const float* cfg_b2     = (const float*)d_in[10];
    const float* n1w        = (const float*)d_in[11];
    const float* n2w        = (const float*)d_in[12];
    const float* qkvw       = (const float*)d_in[13];
    const float* projw      = (const float*)d_in[14];
    const float* projb      = (const float*)d_in[15];
    const float* qnw        = (const float*)d_in[16];
    const float* knw        = (const float*)d_in[17];
    const float* w1         = (const float*)d_in[18];
    const float* w2         = (const float*)d_in[19];
    const float* w3         = (const float*)d_in[20];
    const float* adaw       = (const float*)d_in[21];
    const float* adab       = (const float*)d_in[22];
    const float* fin_ada_w  = (const float*)d_in[23];
    const float* fin_ada_b  = (const float*)d_in[24];
    const float* fin_lin_w  = (const float*)d_in[25];
    const float* fin_lin_b  = (const float*)d_in[26];
    float* outp = (float*)d_out;

    const int Bm = 8, C = 1024, Sw = 2730, Swp = 2752, M = 8 * 256;

    // ---- workspace layout ----
    char* wsb = (char*)d_ws;
    size_t off = 0;
    auto allocf = [&](size_t n) { float* p = (float*)(wsb + off); off += n * 4; return p; };
    auto alloch = [&](size_t n) { unsigned short* p = (unsigned short*)(wsb + off);
                                  off += ((n * 2 + 15) & ~(size_t)15); return p; };
    float* silu_c = allocf((size_t)Bm * C);
    float* mods   = allocf((size_t)4 * Bm * 6 * C);
    float* finmod = allocf((size_t)Bm * 2 * C);
    float* xc     = allocf((size_t)M * C);
    float* big0   = allocf((size_t)M * 3 * C);        // qkv f32 / h1 f32 / proj-out / final xm
    float* big1f  = allocf((size_t)M * Sw);           // h3 f32 / w2-out
    unsigned short* xmbf  = alloch((size_t)M * C);
    unsigned short* qbf   = alloch((size_t)M * C);
    unsigned short* kbf   = alloch((size_t)M * C);
    unsigned short* vbf   = alloch((size_t)M * C);
    unsigned short* obf   = alloch((size_t)M * C);
    unsigned short* h12bf = alloch((size_t)M * Swp);
    unsigned short* wq    = alloch((size_t)3 * C * C);
    unsigned short* wp    = alloch((size_t)C * C);
    unsigned short* ww1   = alloch((size_t)Sw * C);
    unsigned short* ww3   = alloch((size_t)Sw * C);
    unsigned short* ww2   = alloch((size_t)C * Swp);

    patch_embed_kernel<<<M, 256, 0, stream>>>(x, patch_w, patch_b, pos_embed, xc);
    cond_kernel<<<Bm, 256, 0, stream>>>(y, cfg_scale, class_emb, cfg_w1, cfg_b1,
                                        cfg_w2, cfg_b2, silu_c);
    {
        dim3 g(cdiv(6 * C, 256), 4 * Bm);
        vecmat_kernel<<<g, 256, 0, stream>>>(silu_c, adaw, adab, mods, Bm, 6 * C, C);
        dim3 gf(cdiv(2 * C, 256), Bm);
        vecmat_kernel<<<gf, 256, 0, stream>>>(silu_c, fin_ada_w, fin_ada_b, finmod, Bm, 2 * C, C);
    }

    for (int l = 0; l < 4; ++l) {
        const float* mods_l = mods + (size_t)l * Bm * 6 * C;
        // ---- attention branch ----
        {
            long tot = (long)3 * C * C;
            cvt_bf16_kernel<<<cdiv(tot, 256), 256, 0, stream>>>(qkvw + (size_t)l * tot, wq, C, C, tot);
        }
        modulate_rms_kernel<<<M, 256, 0, stream>>>(xc, n1w + l * C, mods_l, 0, C, xmbf);
        {
            dim3 g(3 * C / 64, M / 64);
            gemm_bf16_kernel<<<g, 128, 0, stream>>>(xmbf, wq, nullptr, big0, M, 3 * C, C, C, C);
        }
        qkv_post_kernel<<<M, 256, 0, stream>>>(big0, qnw + l * 64, knw + l * 64, qbf, kbf, vbf);
        attention_kernel<<<Bm * 16, 128, ATT_SMEM, stream>>>(qbf, kbf, vbf, obf);
        {
            long tot = (long)C * C;
            cvt_bf16_kernel<<<cdiv(tot, 256), 256, 0, stream>>>(projw + (size_t)l * tot, wp, C, C, tot);
            dim3 g(C / 64, M / 64);
            gemm_bf16_kernel<<<g, 128, 0, stream>>>(obf, wp, projb + (size_t)l * C, big0, M, C, C, C, C);
        }
        residual_kernel<<<cdiv((long)M * C, 256), 256, 0, stream>>>(xc, big0, mods_l, 2 * C, M * C);
        // ---- SwiGLU MLP branch ----
        modulate_rms_kernel<<<M, 256, 0, stream>>>(xc, n2w + l * C, mods_l, 3 * C, 4 * C, xmbf);
        {
            long tot = (long)Sw * C;
            cvt_bf16_kernel<<<cdiv(tot, 256), 256, 0, stream>>>(w1 + (size_t)l * tot, ww1, C, C, tot);
            cvt_bf16_kernel<<<cdiv(tot, 256), 256, 0, stream>>>(w3 + (size_t)l * tot, ww3, C, C, tot);
            dim3 g(cdiv(Sw, 64), M / 64);
            gemm_bf16_kernel<<<g, 128, 0, stream>>>(xmbf, ww1, nullptr, big0, M, Sw, C, C, C);
            gemm_bf16_kernel<<<g, 128, 0, stream>>>(xmbf, ww3, nullptr, big1f, M, Sw, C, C, C);
        }
        swiglu_bf16_kernel<<<cdiv((long)M * Swp, 256), 256, 0, stream>>>(
            big0, big1f, h12bf, Sw, Swp, (long)M * Swp);
        {
            long tot = (long)C * Swp;
            cvt_bf16_kernel<<<cdiv(tot, 256), 256, 0, stream>>>(w2 + (size_t)l * C * Sw, ww2, Sw, Swp, tot);
            dim3 g(C / 64, M / 64);
            gemm_bf16_kernel<<<g, 128, 0, stream>>>(h12bf, ww2, nullptr, big1f, M, C, Swp, Swp, Swp);
        }
        residual_kernel<<<cdiv((long)M * C, 256), 256, 0, stream>>>(xc, big1f, mods_l, 5 * C, M * C);
    }

    final_mod_kernel<<<M, 256, 0, stream>>>(xc, finmod, big0);
    final_linear_kernel<<<M, 256, 0, stream>>>(big0, fin_lin_w, fin_lin_b, outp);
}
